// PPO_34282428956970
// MI455X (gfx1250) — matrix operand order repaired and checked by reading, not compiled, not run on hardware
//
#include <hip/hip_runtime.h>
#include <hip/hip_bf16.h>

typedef __attribute__((ext_vector_type(16))) __bf16 v16bf;
typedef __attribute__((ext_vector_type(8)))  float  v8f;

#define WAVES 4
#define BLOCK (WAVES * 32)
#define ASTR  208          // halves per staged A row: 192 + 16 pad -> 416B, 16B aligned
#define FDIM  64
#define MNBR  32
#define KDIM  192

#define LOG2E 1.4426950408889634f
#define LN2   0.6931471805599453f

__device__ __forceinline__ unsigned short f32_to_bf16(float x) {
    unsigned u = __float_as_uint(x);
    u += 0x7FFFu + ((u >> 16) & 1u);          // round-to-nearest-even
    return (unsigned short)(u >> 16);
}
// RNE-round both floats, pack high halves with one v_perm_b32
__device__ __forceinline__ unsigned pack2_bf16(float lo, float hi) {
    unsigned ul = __float_as_uint(lo), uh = __float_as_uint(hi);
    ul += 0x7FFFu + ((ul >> 16) & 1u);
    uh += 0x7FFFu + ((uh >> 16) & 1u);
    return __builtin_amdgcn_perm(uh, ul, 0x07060302u);  // {uh[31:16], ul[31:16]}
}
// hardware-transcendental activations: v_exp_f32 / v_log_f32 / v_rcp_f32
__device__ __forceinline__ float fexp(float x)     { return __builtin_amdgcn_exp2f(x * LOG2E); }
__device__ __forceinline__ float fsigmoid(float x) { return __builtin_amdgcn_rcpf(1.0f + fexp(-x)); }
__device__ __forceinline__ float fsoftplus(float x) {
    return fmaxf(x, 0.0f) + LN2 * __builtin_amdgcn_logf(1.0f + fexp(-fabsf(x)));
}

union FragAB { v16bf v; float4 q[2]; };

// Repack W (128x192 f32, row-major [o][k]) into bf16 B-matrix fragments:
// frag = nt*6 + kt (48 frags), each frag = 32 lanes x 16 halves (1 KB), so the
// GEMM kernel loads each lane's 32 bytes contiguously & fully coalesced.
// B layout (32x16 bf16): lane = column n; lanes 0-15 hold K 0..7,16..23, lanes 16-31 hold K 8..15,24..31.
__global__ void wpack_kernel(const float* __restrict__ W, unsigned short* __restrict__ wb) {
    int i = blockIdx.x * blockDim.x + threadIdx.x;
    if (i >= 48 * 512) return;
    int frag = i >> 9;
    int lane = (i >> 4) & 31;
    int e    = i & 15;
    int nt = frag / 6, kt = frag % 6;
    int o = nt * 16 + (lane & 15);
    int k = kt * 32 + ((lane < 16) ? 0 : 8) + (e & 7) + ((e >> 3) << 4);
    wb[i] = f32_to_bf16(W[o * KDIM + k]);
}

__global__ void __launch_bounds__(BLOCK)
cgcnn_wmma_kernel(const float* __restrict__ node, const float* __restrict__ edge,
                  const int* __restrict__ idx, const unsigned short* __restrict__ wb,
                  const float* __restrict__ bias, const float* __restrict__ alpha_p,
                  float* __restrict__ out, int Ntot) {
    __shared__ __align__(16) unsigned short Ast[WAVES * MNBR * ASTR];
    const int lane = threadIdx.x & 31;
    const int wave = threadIdx.x >> 5;
    const int gw   = blockIdx.x * WAVES + wave;
    const int tw   = gridDim.x * WAVES;
    const float alpha = alpha_p[0];
    const float4*  wq    = (const float4*)wb;
    const float2*  node2 = (const float2*)node;   // 32 float2 per node row
    const float2*  edge2 = (const float2*)edge;
    const int c  = lane & 15;
    const int hi = (lane < 16) ? 0 : 1;

    const int iters = (Ntot + tw - 1) / tw;
    for (int it = 0; it < iters; ++it) {
        const int n = gw + it * tw;
        const bool active = (n < Ntot);          // wave-uniform branch: EXEC stays full
        unsigned vote = 0u;
        if (active) {
            // ---- stage A = [self(64) | nbr(64) | edge(64)] as packed bf16 pairs ----
            // lane handles features 2*lane, 2*lane+1 of each 64-wide chunk:
            // one global_load_b64 + one packed cvt + one ds_store_b32 per chunk per row.
            float2 sv = node2[(size_t)n * 32 + lane];
            const unsigned selfp = pack2_bf16(sv.x, sv.y);
            int j = idx[n * MNBR + lane];                  // lane l <-> neighbor m=l
            vote  = (unsigned)__ballot(j >= 0);            // validity mask, 1 bit per m
            unsigned* aw32 = (unsigned*)&Ast[wave * MNBR * ASTR];
            for (int m = 0; m < MNBR; ++m) {
                int jm = __shfl(j, m, 32);
                int js = (jm < 0) ? 0 : jm;                // clip(-1 -> 0); masked later
                float2 nv = node2[(size_t)js * 32 + lane];
                float2 ev = edge2[(size_t)(n * MNBR + m) * 32 + lane];
                unsigned* row = aw32 + m * (ASTR / 2);
                row[lane]      = selfp;
                row[32 + lane] = pack2_bf16(nv.x, nv.y);
                row[64 + lane] = pack2_bf16(ev.x, ev.y);
            }
        }
        // Wave-private LDS region + in-order DS pipeline per wave: only a
        // compiler-visible ordering point is needed, not a block barrier.
        __builtin_amdgcn_fence(__ATOMIC_ACQ_REL, "wavefront");
        if (active) {
            // per-lane neighbor validity masks, computed once per node
            float km0[8], km1[8];
            #pragma unroll
            for (int r = 0; r < 8; ++r) {
                int m0 = hi ? (r + 8) : r;
                km0[r] = ((vote >> m0) & 1u)        ? 1.0f : 0.0f;   // mt=0 rows
                km1[r] = ((vote >> (16 + m0)) & 1u) ? 1.0f : 0.0f;   // mt=1 rows
            }
            // ---- load all 12 A fragments (mt 0..1, kt 0..5), ISA 16-bit A layout ----
            // lane<16: e0..7 -> K=kt*32+e, e8..15 -> K=kt*32+16+(e-8); lane>=16: +8
            FragAB A[12];
            const unsigned short* aw = &Ast[wave * MNBR * ASTR];
            #pragma unroll
            for (int mt = 0; mt < 2; ++mt)
                #pragma unroll
                for (int kt = 0; kt < 6; ++kt) {
                    int h = (mt * 16 + c) * ASTR + kt * 32 + hi * 8;
                    const float4* p = (const float4*)(aw + h);     // 16B aligned
                    A[mt * 6 + kt].q[0] = p[0];
                    A[mt * 6 + kt].q[1] = p[2];                    // +32 bytes
                }
            #pragma unroll
            for (int nt = 0; nt < 4; ++nt) {
                const float bf_ = bias[nt * 16 + c];        // filter bias (o = nt*16+c)
                const float bc_ = bias[64 + nt * 16 + c];   // core bias
                v8f F0, F1, C0, C1;
                #pragma unroll
                for (int r = 0; r < 8; ++r) { F0[r] = bf_; F1[r] = bf_; C0[r] = bc_; C1[r] = bc_; }
                #pragma unroll
                for (int kt = 0; kt < 6; ++kt) {
                    FragAB Bf, Bc;
                    const int fF = nt * 6 + kt;            // filter column block
                    const int fC = (nt + 4) * 6 + kt;      // core   column block
                    const int qi = lane * 2;
                    Bf.q[0] = wq[fF * 64 + qi];  Bf.q[1] = wq[fF * 64 + qi + 1];
                    Bc.q[0] = wq[fC * 64 + qi];  Bc.q[1] = wq[fC * 64 + qi + 1];
                    F0 = __builtin_amdgcn_wmma_f32_16x16x32_bf16(false, A[kt].v,     false, Bf.v, (short)0, F0, false, false);
                    F1 = __builtin_amdgcn_wmma_f32_16x16x32_bf16(false, A[6 + kt].v, false, Bf.v, (short)0, F1, false, false);
                    C0 = __builtin_amdgcn_wmma_f32_16x16x32_bf16(false, A[kt].v,     false, Bc.v, (short)0, C0, false, false);
                    C1 = __builtin_amdgcn_wmma_f32_16x16x32_bf16(false, A[6 + kt].v, false, Bc.v, (short)0, C1, false, false);
                }
                // ---- epilogue: masked sigmoid(filter)*softplus(core), sum over m ----
                // C layout: VGPR r -> row M=r (lanes 0-15) / M=r+8 (lanes 16-31); lane&15 -> N
                float s = 0.0f;
                #pragma unroll
                for (int r = 0; r < 8; ++r) {
                    s += km0[r] * fsigmoid(F0[r]) * fsoftplus(C0[r]);
                    s += km1[r] * fsigmoid(F1[r]) * fsoftplus(C1[r]);
                }
                s += __shfl_xor(s, 16, 32);                 // fold lane halves (m 0-7 + 8-15)
                if (lane < 16) {
                    int o = nt * 16 + lane;
                    out[n * FDIM + o] = fsoftplus(alpha * node[n * FDIM + o] + s);
                }
            }
        }
        // order this iteration's fragment loads before next iteration's staging stores
        __builtin_amdgcn_fence(__ATOMIC_ACQ_REL, "wavefront");
    }
}

extern "C" void kernel_launch(void* const* d_in, const int* in_sizes, int n_in,
                              void* d_out, int out_size, void* d_ws, size_t ws_size,
                              hipStream_t stream) {
    const float* node  = (const float*)d_in[0];   // (N, 64) f32
    const float* edge  = (const float*)d_in[1];   // (N, 32, 64) f32
    const int*   idx   = (const int*)d_in[2];     // (N, 32) i32
    const float* W     = (const float*)d_in[3];   // (128, 192) f32
    const float* b     = (const float*)d_in[4];   // (128,) f32
    const float* alpha = (const float*)d_in[5];   // scalar f32
    float* out = (float*)d_out;                   // (N, 64) f32
    const int Ntot = in_sizes[0] / FDIM;

    unsigned short* wb = (unsigned short*)d_ws;   // 48 KB bf16 fragment-packed W
    wpack_kernel<<<(48 * 512 + 255) / 256, 256, 0, stream>>>(W, wb);

    const int blocks = 2048;                      // 8192 waves, ~7 nodes/wave grid-stride
    cgcnn_wmma_kernel<<<blocks, BLOCK, 0, stream>>>(node, edge, idx, wb, b, alpha, out, Ntot);

    (void)n_in; (void)out_size; (void)ws_size;
}
